// BitNetAttention_76424648065996
// MI455X (gfx1250) — 8-bit matrix operand order repaired and checked by reading, compiled, not run on hardware
//
#include <hip/hip_runtime.h>
#include <hip/hip_bf16.h>

#define HIDDEN 2048
#define SEQ    2048
#define NB     2
#define NH     16
#define HD     128
#define MROWS  (NB * SEQ)
#define WELEM  (HIDDEN * HIDDEN)

typedef __attribute__((ext_vector_type(8)))  int      v8i;
typedef __attribute__((ext_vector_type(8)))  float    v8f;
typedef __attribute__((ext_vector_type(16))) _Float16 v16h;
typedef __attribute__((ext_vector_type(8)))  _Float16 h8;
typedef __attribute__((ext_vector_type(2)))  int      i2;
typedef __attribute__((ext_vector_type(4)))  int      i4;
typedef __attribute__((ext_vector_type(4)))  unsigned int u32x4;
typedef __attribute__((ext_vector_type(8)))  int      i32x8;
typedef __attribute__((ext_vector_type(4)))  int      i32x4;

union FragI  { v8i v; int i[8]; i2 d2[4]; i4 d4[2]; };
union Frag16 { v16h v; h8 h[2]; };

#if __has_builtin(__builtin_amdgcn_tensor_load_to_lds)
#define HAVE_TDM 1
#endif

#ifdef HAVE_TDM
__device__ __forceinline__ void tdm_load_2d(const void* gsrc, unsigned int laddr,
                                            int dsLog2, unsigned int tensorD0,
                                            unsigned int tensorD1, unsigned int tileD0,
                                            unsigned int tileD1, unsigned int stride0) {
  const unsigned long long ga = (unsigned long long)gsrc;
  u32x4 g0;
  g0[0] = 1u;
  g0[1] = laddr;
  g0[2] = (unsigned int)(ga & 0xFFFFFFFFu);
  g0[3] = (unsigned int)((ga >> 32) & 0x01FFFFFFu)
          | 0x80000000u;
  i32x8 g1;
  g1[0] = dsLog2 << 16;
  g1[1] = (int)((tensorD0 & 0xFFFFu) << 16);
  g1[2] = (int)(((tensorD0 >> 16) & 0xFFFFu) | ((tensorD1 & 0xFFFFu) << 16));
  g1[3] = (int)(((tensorD1 >> 16) & 0xFFFFu) | ((tileD0 & 0xFFFFu) << 16));
  g1[4] = (int)(tileD1 & 0xFFFFu);
  g1[5] = (int)stride0;
  g1[6] = 0;
  g1[7] = 0;
  i32x4 gz = {0, 0, 0, 0};
#if __clang_major__ >= 23
  i32x8 gz8 = {0, 0, 0, 0, 0, 0, 0, 0};
  __builtin_amdgcn_tensor_load_to_lds(g0, g1, gz, gz, gz8, 0);
#else
  __builtin_amdgcn_tensor_load_to_lds(g0, g1, gz, gz, 0);
#endif
}
#endif

__global__ void k_zero4(float* p) {
  if (threadIdx.x < 4) p[threadIdx.x] = 0.0f;
}

__global__ void k_abssum(const float* __restrict__ w, float* __restrict__ out, int n) {
  __shared__ float red[256];
  float s = 0.0f;
  for (int i = blockIdx.x * blockDim.x + threadIdx.x; i < n; i += gridDim.x * blockDim.x)
    s += fabsf(w[i]);
  red[threadIdx.x] = s;
  __syncthreads();
  for (int st = 128; st > 0; st >>= 1) {
    if ((int)threadIdx.x < st) red[threadIdx.x] += red[threadIdx.x + st];
    __syncthreads();
  }
  if (threadIdx.x == 0) atomicAdd(out, red[0]);
}

__global__ void k_quant_w(const float* __restrict__ w, signed char* __restrict__ wq,
                          const float* __restrict__ sums, int idx, int n) {
  const float s   = sums[idx] * (1.0f / (float)WELEM) + 1e-5f;
  const float inv = 1.0f / s;
  for (int i = blockIdx.x * blockDim.x + threadIdx.x; i < n; i += gridDim.x * blockDim.x) {
    float q = rintf(w[i] * inv);
    q = fminf(1.0f, fmaxf(-1.0f, q));
    wq[i] = (signed char)q;
  }
}

__global__ void k_quant_act(const float* __restrict__ x, signed char* __restrict__ xq,
                            float* __restrict__ gscale) {
  __shared__ float red[256];
  const int row = blockIdx.x;
  const float* xr = x + (size_t)row * HIDDEN;
  float vals[8];
  float m = 0.0f;
#pragma unroll
  for (int i = 0; i < 8; ++i) {
    vals[i] = xr[threadIdx.x + i * 256];
    m = fmaxf(m, fabsf(vals[i]));
  }
  red[threadIdx.x] = m;
  __syncthreads();
  for (int st = 128; st > 0; st >>= 1) {
    if ((int)threadIdx.x < st)
      red[threadIdx.x] = fmaxf(red[threadIdx.x], red[threadIdx.x + st]);
    __syncthreads();
  }
  const float g   = red[0] + 1e-5f;
  const float inv = 127.0f / g;
#pragma unroll
  for (int i = 0; i < 8; ++i) {
    float q = rintf(vals[i] * inv);
    q = fminf(127.0f, fmaxf(-127.0f, q));
    xq[(size_t)row * HIDDEN + threadIdx.x + i * 256] = (signed char)q;
  }
  if (threadIdx.x == 0) gscale[row] = g * (1.0f / 127.0f);
}

__global__ __launch_bounds__(256) void k_gemm_i8(
    const signed char* __restrict__ aq, const float* __restrict__ arowscale,
    const signed char* __restrict__ wqAll, const float* __restrict__ wsums,
    int wBase, _Float16* __restrict__ o0, _Float16* __restrict__ o1,
    _Float16* __restrict__ o2, float* __restrict__ oF) {
  const int lane = threadIdx.x & 31;
  const int wave = threadIdx.x >> 5;
  const int r    = lane & 15;
  const int half = lane >> 4;
  const int z    = blockIdx.z;
  const int mBase = blockIdx.y * 32 + (wave & 1) * 16;
  const int nBase = blockIdx.x * 256 + (wave >> 1) * 64;

  const signed char* __restrict__ wq = wqAll + (size_t)(wBase + z) * WELEM;

  FragI acc[4];
#pragma unroll
  for (int t = 0; t < 4; ++t) acc[t].v = (v8i){0, 0, 0, 0, 0, 0, 0, 0};

  const int h16 = half * 16;
  const int h8o = half * 8;

#ifdef HAVE_TDM
  __shared__ signed char ldsA[2][32 * 128];
  __shared__ signed char ldsB[2][256 * 128];

  const signed char* aTile = aq + (size_t)(blockIdx.y * 32) * HIDDEN;
  const signed char* bTile = wq + (size_t)(blockIdx.x * 256) * HIDDEN;
  const int rowA = (wave & 1) * 16 + r;
  const int rowB = (wave >> 1) * 64;

  if (wave == 0) {
    tdm_load_2d(aTile, (unsigned int)(size_t)(void*)&ldsA[0][0], 0,
                HIDDEN, MROWS, 128, 32, HIDDEN);
    tdm_load_2d(bTile, (unsigned int)(size_t)(void*)&ldsB[0][0], 0,
                HIDDEN, HIDDEN, 128, 256, HIDDEN);
  }

  for (int kc = 0; kc < HIDDEN; kc += 128) {
    const int buf = (kc >> 7) & 1;
    __syncthreads();
    if (wave == 0) {
      if (kc + 128 < HIDDEN) {
        tdm_load_2d(aTile + kc + 128, (unsigned int)(size_t)(void*)&ldsA[buf ^ 1][0],
                    0, HIDDEN, MROWS, 128, 32, HIDDEN);
        tdm_load_2d(bTile + kc + 128, (unsigned int)(size_t)(void*)&ldsB[buf ^ 1][0],
                    0, HIDDEN, HIDDEN, 128, 256, HIDDEN);
        __builtin_amdgcn_s_wait_tensorcnt(2);
      } else {
        __builtin_amdgcn_s_wait_tensorcnt(0);
      }
    }
    __syncthreads();

#pragma unroll
    for (int ks = 0; ks < 128; ks += 64) {
      FragI a;
      const signed char* ar = &ldsA[buf][rowA * 128 + ks];
      a.d2[0] = *(const i2*)(ar + h16);
      a.d2[1] = *(const i2*)(ar + h16 + 8);
      a.d2[2] = *(const i2*)(ar + h16 + 32);
      a.d2[3] = *(const i2*)(ar + h16 + 40);
#pragma unroll
      for (int t = 0; t < 4; ++t) {
        FragI b;
        const signed char* br = &ldsB[buf][(rowB + t * 16 + r) * 128 + ks];
        b.d4[0] = *(const i4*)(br + h16);
        b.d4[1] = *(const i4*)(br + h16 + 32);
        acc[t].v = __builtin_amdgcn_wmma_i32_16x16x64_iu8(
            true, a.v, true, b.v, acc[t].v, false, false);
      }
    }
  }
#else
  const signed char* __restrict__ arow = aq + (size_t)(mBase + r) * HIDDEN;
  const signed char* bcol[4];
#pragma unroll
  for (int t = 0; t < 4; ++t)
    bcol[t] = wq + (size_t)(nBase + t * 16 + r) * HIDDEN;

  auto loadAB = [&](FragI& A, FragI* B, int kc) {
    A.d2[0] = *(const i2*)(arow + kc + h16);
    A.d2[1] = *(const i2*)(arow + kc + h16 + 8);
    A.d2[2] = *(const i2*)(arow + kc + h16 + 32);
    A.d2[3] = *(const i2*)(arow + kc + h16 + 40);
#pragma unroll
    for (int t = 0; t < 4; ++t) {
      B[t].d4[0] = *(const i4*)(bcol[t] + kc + h16);
      B[t].d4[1] = *(const i4*)(bcol[t] + kc + h16 + 32);
    }
  };
  FragI a[2];
  FragI bb[2][4];
  loadAB(a[0], bb[0], 0);
  int pb = 0;
  for (int kc = 0; kc < HIDDEN; kc += 64, pb ^= 1) {
    if (kc + 64 < HIDDEN) {
      __builtin_prefetch(arow + kc + 192, 0, 3);
      loadAB(a[pb ^ 1], bb[pb ^ 1], kc + 64);
    }
#pragma unroll
    for (int t = 0; t < 4; ++t)
      acc[t].v = __builtin_amdgcn_wmma_i32_16x16x64_iu8(
          true, a[pb].v, true, bb[pb][t].v, acc[t].v, false, false);
  }
#endif

  const float sw = wsums[wBase + z] * (1.0f / (float)WELEM) + 1e-5f;
  _Float16* __restrict__ oh = (z == 0) ? o0 : (z == 1) ? o1 : o2;

#pragma unroll
  for (int g = 0; g < 8; ++g) {
    const int   m  = mBase + g + half * 8;
    const float sc = sw * arowscale[m];
#pragma unroll
    for (int t = 0; t < 4; ++t) {
      const int   n  = nBase + t * 16 + r;
      const float vv = (float)acc[t].i[g] * sc;
      if (oF) oF[(size_t)m * HIDDEN + n] = vv;
      else    oh[(size_t)m * HIDDEN + n] = (_Float16)vv;
    }
  }
}

__global__ void k_rope(_Float16* __restrict__ qh, _Float16* __restrict__ kh) {
  const size_t idx = (size_t)blockIdx.x * blockDim.x + threadIdx.x;
  const int    d   = (int)(idx & 63);
  const size_t rh  = idx >> 6;
  const int    h   = (int)(rh & (NH - 1));
  const size_t m   = rh >> 4;
  const int    s   = (int)(m & (SEQ - 1));
  const float invf = __expf((float)d * -0.14391156516f);
  float sn, cs;
  __sincosf((float)s * invf, &sn, &cs);
  const size_t base = m * HIDDEN + (size_t)h * HD + d;
  float q1 = (float)qh[base], q2 = (float)qh[base + 64];
  qh[base]      = (_Float16)(q1 * cs - q2 * sn);
  qh[base + 64] = (_Float16)(q2 * cs + q1 * sn);
  float k1 = (float)kh[base], k2 = (float)kh[base + 64];
  kh[base]      = (_Float16)(k1 * cs - k2 * sn);
  kh[base + 64] = (_Float16)(k2 * cs + k1 * sn);
}

__global__ __launch_bounds__(128) void k_attn(
    const _Float16* __restrict__ qh, const _Float16* __restrict__ kh,
    const _Float16* __restrict__ vh, float* __restrict__ ctx) {
  __shared__ _Float16 ldsK[32 * HD];
  __shared__ _Float16 ldsVt[HD * 32];
  __shared__ _Float16 ldsP[4][16 * 32];

  const int lane = threadIdx.x & 31;
  const int wave = threadIdx.x >> 5;
  const int r    = lane & 15;
  const int half = lane >> 4;
  const int h    = blockIdx.y;
  const int b    = blockIdx.z;
  const size_t rowBase = (size_t)b * SEQ;
  const int qRow0 = blockIdx.x * 64 + wave * 16;

  Frag16 qa[4];
  {
    const _Float16* qrow = qh + (rowBase + qRow0 + r) * (size_t)HIDDEN + h * HD;
#pragma unroll
    for (int c = 0; c < 4; ++c) {
      qa[c].h[0] = *(const h8*)(qrow + c * 32 + half * 8);
      qa[c].h[1] = *(const h8*)(qrow + c * 32 + half * 8 + 16);
    }
  }

  float mrow[8], lrow[8];
  v8f acc[8];
#pragma unroll
  for (int g = 0; g < 8; ++g) { mrow[g] = -3.0e30f; lrow[g] = 0.0f; }
#pragma unroll
  for (int t = 0; t < 8; ++t) acc[t] = (v8f){0, 0, 0, 0, 0, 0, 0, 0};

  const float scale = 0.08838834764831845f;

  for (int kv = 0; kv < SEQ; kv += 32) {
    __syncthreads();

#ifdef HAVE_TDM
    if (wave == 0) {
      tdm_load_2d(kh + (rowBase + kv) * (size_t)HIDDEN + h * HD,
                  (unsigned int)(size_t)(void*)&ldsK[0],
                  1, HIDDEN, MROWS, HD, 32, HIDDEN);
    }
#endif

    {
      const int rL   = threadIdx.x >> 2;
      const int dOff = (threadIdx.x & 3) * 32;
      const _Float16* vsrc = vh + (rowBase + kv + rL) * (size_t)HIDDEN + h * HD + dOff;
#ifndef HAVE_TDM
      const _Float16* ksrc = kh + (rowBase + kv + rL) * (size_t)HIDDEN + h * HD + dOff;
#endif
#pragma unroll
      for (int j = 0; j < 4; ++j) {
#ifndef HAVE_TDM
        h8 kd = *(const h8*)(ksrc + j * 8);
        *(h8*)&ldsK[rL * HD + dOff + j * 8] = kd;
#endif
        h8 vd = *(const h8*)(vsrc + j * 8);
#pragma unroll
        for (int e = 0; e < 8; ++e)
          ldsVt[(dOff + j * 8 + e) * 32 + rL] = vd[e];
      }
    }

#ifdef HAVE_TDM
    if (wave == 0) __builtin_amdgcn_s_wait_tensorcnt(0);
#endif
    __syncthreads();

    v8f sT[2];
#pragma unroll
    for (int j = 0; j < 2; ++j) {
      v8f s = (v8f){0, 0, 0, 0, 0, 0, 0, 0};
      const _Float16* krow = &ldsK[(j * 16 + r) * HD];
#pragma unroll
      for (int c = 0; c < 4; ++c) {
        Frag16 kb;
        kb.h[0] = *(const h8*)(krow + c * 32 + half * 8);
        kb.h[1] = *(const h8*)(krow + c * 32 + half * 8 + 16);
        s = __builtin_amdgcn_wmma_f32_16x16x32_f16(false, qa[c].v, false, kb.v,
                                                   (short)0, s, false, false);
      }
      sT[j] = s;
    }

#pragma unroll
    for (int g = 0; g < 8; ++g) {
      float v0 = sT[0][g] * scale;
      float v1 = sT[1][g] * scale;
      float mx = fmaxf(v0, v1);
#pragma unroll
      for (int off = 1; off < 16; off <<= 1)
        mx = fmaxf(mx, __shfl_xor(mx, off, 32));
      const float mNew = fmaxf(mrow[g], mx);
      const float corr = __expf(mrow[g] - mNew);
      mrow[g] = mNew;
      const float p0 = __expf(v0 - mNew);
      const float p1 = __expf(v1 - mNew);
      float rs = p0 + p1;
#pragma unroll
      for (int off = 1; off < 16; off <<= 1)
        rs += __shfl_xor(rs, off, 32);
      lrow[g] = lrow[g] * corr + rs;
#pragma unroll
      for (int t = 0; t < 8; ++t) acc[t][g] *= corr;
      const int rr = g + half * 8;
      ldsP[wave][rr * 32 + r]      = (_Float16)p0;
      ldsP[wave][rr * 32 + 16 + r] = (_Float16)p1;
    }

    Frag16 pa;
    pa.h[0] = *(const h8*)(&ldsP[wave][r * 32 + half * 8]);
    pa.h[1] = *(const h8*)(&ldsP[wave][r * 32 + half * 8 + 16]);
#pragma unroll
    for (int t = 0; t < 8; ++t) {
      Frag16 vb;
      const _Float16* vcol = &ldsVt[(t * 16 + r) * 32];
      vb.h[0] = *(const h8*)(vcol + half * 8);
      vb.h[1] = *(const h8*)(vcol + half * 8 + 16);
      acc[t] = __builtin_amdgcn_wmma_f32_16x16x32_f16(false, pa.v, false, vb.v,
                                                      (short)0, acc[t], false, false);
    }
  }

#pragma unroll
  for (int g = 0; g < 8; ++g) {
    const int   rr   = g + half * 8;
    const float invl = 1.0f / lrow[g];
    float* crow = ctx + (rowBase + qRow0 + rr) * (size_t)HIDDEN + h * HD;
#pragma unroll
    for (int t = 0; t < 8; ++t)
      crow[t * 16 + r] = acc[t][g] * invl;
  }
}

extern "C" void kernel_launch(void* const* d_in, const int* in_sizes, int n_in,
                              void* d_out, int out_size, void* d_ws, size_t ws_size,
                              hipStream_t stream) {
  (void)in_sizes; (void)n_in; (void)out_size; (void)ws_size;
  const float* x = (const float*)d_in[0];
  const float* w[4] = {(const float*)d_in[1], (const float*)d_in[2],
                       (const float*)d_in[3], (const float*)d_in[4]};
  float* out = (float*)d_out;
  char*  ws  = (char*)d_ws;

  float* sums = (float*)ws;
  size_t off = 256;
  signed char* wq = (signed char*)(ws + off); off += 4ull * WELEM;
  signed char* xq = (signed char*)(ws + off); off += (size_t)MROWS * HIDDEN;
  float* ascale   = (float*)(ws + off);       off += (size_t)MROWS * 4;
  _Float16* qh    = (_Float16*)(ws + off);    off += (size_t)MROWS * HIDDEN * 2;
  _Float16* kh    = (_Float16*)(ws + off);    off += (size_t)MROWS * HIDDEN * 2;
  _Float16* vh    = (_Float16*)(ws + off);    off += (size_t)MROWS * HIDDEN * 2;
  float* ctx      = (float*)(ws + off);       off += (size_t)MROWS * HIDDEN * 4;

  k_zero4<<<1, 32, 0, stream>>>(sums);
  for (int i = 0; i < 4; ++i)
    k_abssum<<<256, 256, 0, stream>>>(w[i], sums + i, WELEM);
  for (int i = 0; i < 4; ++i)
    k_quant_w<<<1024, 256, 0, stream>>>(w[i], wq + (size_t)i * WELEM, sums, i, WELEM);

  k_quant_act<<<MROWS, 256, 0, stream>>>(x, xq, ascale);

  k_gemm_i8<<<dim3(8, 128, 3), 256, 0, stream>>>(xq, ascale, wq, sums, 0,
                                                 qh, kh, vh, nullptr);
  k_rope<<<16384, 256, 0, stream>>>(qh, kh);
  k_attn<<<dim3(SEQ / 64, NH, NB), 128, 0, stream>>>(qh, kh, vh, ctx);

  k_quant_act<<<MROWS, 256, 0, stream>>>(ctx, xq, ascale);
  k_gemm_i8<<<dim3(8, 128, 1), 256, 0, stream>>>(xq, ascale, wq, sums, 3,
                                                 nullptr, nullptr, nullptr, out);
}
